// DotAttention_15187004358886
// MI455X (gfx1250) — matrix operand order repaired and checked by reading, not compiled, not run on hardware
//
#include <hip/hip_runtime.h>
#include <hip/hip_bf16.h>

typedef __attribute__((ext_vector_type(16))) __bf16 v16bf;
typedef __attribute__((ext_vector_type(8)))  float  v8f;
typedef int v4i __attribute__((vector_size(16)));
typedef __attribute__((address_space(1))) v4i* gas_v4i_ptr;
typedef __attribute__((address_space(3))) v4i* las_v4i_ptr;

#define NEG_INF_F (-1e30f)

#if __has_builtin(__builtin_amdgcn_global_load_async_to_lds_b128) && \
    __has_builtin(__builtin_amdgcn_s_wait_asynccnt)
#define HAS_ASYNC 1
#else
#define HAS_ASYNC 0
#endif

__device__ __forceinline__ unsigned short f2bf(float x) {
    unsigned int u = __float_as_uint(x);
    u += 0x7FFFu + ((u >> 16) & 1u);   // round-to-nearest-even
    return (unsigned short)(u >> 16);
}

// 16B global -> LDS copy: async (ASYNCcnt) when available, else via VGPRs.
__device__ __forceinline__ void cp16(void* lds, const void* g) {
#if HAS_ASYNC
    __builtin_amdgcn_global_load_async_to_lds_b128(
        (gas_v4i_ptr)(g), (las_v4i_ptr)(lds), 0, 0);
#else
    *(uint4*)lds = *(const uint4*)g;
#endif
}

__device__ __forceinline__ void cp_wait() {
#if HAS_ASYNC
    __builtin_amdgcn_s_wait_asynccnt(0);
#endif
}

union FragU {
    v16bf v;
    uint4 q[2];
};

// A-frag (16x32 bf16): lane l -> row m, K = {(l>>4)*8..+7} U {16+(l>>4)*8..+7}
__device__ __forceinline__ v16bf load_fragA(const unsigned short* tile, int row,
                                            int stride, int lane) {
    FragU f;
    const unsigned short* p = tile + (size_t)row * stride + ((lane >> 4) << 3);
    f.q[0] = *(const uint4*)(p);
    f.q[1] = *(const uint4*)(p + 16);
    return f.v;
}

// B-frag (32x16 bf16): lane l -> col n, K = {(l>>4)*8..+7} U {16+(l>>4)*8..+7}
__device__ __forceinline__ v16bf load_fragB(const unsigned short* tile, int col,
                                            int stride, int lane) {
    FragU f;
    const unsigned short* p = tile + (size_t)col * stride + ((lane >> 4) << 3);
    f.q[0] = *(const uint4*)(p);
    f.q[1] = *(const uint4*)(p + 16);
    return f.v;
}

#define WMMA_BF16(A, Bf, C) \
    __builtin_amdgcn_wmma_f32_16x16x32_bf16(false, (A), false, (Bf), (short)0, (C), false, false)

// ---------------------------------------------------------------------------
// Pass 0: fp32 -> bf16 bulk convert (vectorized by 8)
// ---------------------------------------------------------------------------
__global__ __launch_bounds__(256)
void cvt_bf16_kernel(const float* __restrict__ in, unsigned short* __restrict__ out,
                     size_t n) {
    size_t i = ((size_t)blockIdx.x * 256 + threadIdx.x) * 8;
    const size_t stride = (size_t)gridDim.x * 256 * 8;
    for (; i + 8 <= n; i += stride) {
        float4 x = *(const float4*)(in + i);
        float4 y = *(const float4*)(in + i + 4);
        unsigned short o[8] = {f2bf(x.x), f2bf(x.y), f2bf(x.z), f2bf(x.w),
                               f2bf(y.x), f2bf(y.y), f2bf(y.z), f2bf(y.w)};
        *(uint4*)(out + i) = *(const uint4*)o;
    }
}

// ---------------------------------------------------------------------------
// Pass 1: scores[b,t,s] = sum_h dec[t,b,h]*enc[s,b,h] (+ length mask)
// 128x128 block tile, 8 waves x (32x64), K-step 32, double-buffered async.
// ---------------------------------------------------------------------------
__global__ __launch_bounds__(256)
void scores_kernel(const unsigned short* __restrict__ encbf,
                   const unsigned short* __restrict__ decbf,
                   const long long* __restrict__ srclen, float* __restrict__ attn,
                   int S, int T, int B, int H) {
    const int s0 = blockIdx.x * 128;
    const int t0 = blockIdx.y * 128;
    const int b  = blockIdx.z;
    const int tid  = threadIdx.x;
    const int lane = tid & 31;
    const int wave = tid >> 5;
    const int wm = wave & 3;   // 32-row group
    const int wn = wave >> 2;  // 64-col group

    __shared__ __align__(16) unsigned short As[2][128 * 32];
    __shared__ __align__(16) unsigned short Bs[2][128 * 32];

    const int srow = tid >> 2;        // 0..63 (and +64)
    const int soff = (tid & 3) << 3;  // 0,8,16,24
    const size_t BH = (size_t)B * H;

    const unsigned short* gA0 = decbf + (size_t)(t0 + srow) * BH + (size_t)b * H + soff;
    const unsigned short* gA1 = gA0 + 64 * BH;
    const unsigned short* gB0 = encbf + (size_t)(s0 + srow) * BH + (size_t)b * H + soff;
    const unsigned short* gB1 = gB0 + 64 * BH;

    unsigned short* const lA0[2] = {&As[0][srow * 32 + soff], &As[1][srow * 32 + soff]};
    unsigned short* const lA1[2] = {lA0[0] + 64 * 32, lA0[1] + 64 * 32};
    unsigned short* const lB0[2] = {&Bs[0][srow * 32 + soff], &Bs[1][srow * 32 + soff]};
    unsigned short* const lB1[2] = {lB0[0] + 64 * 32, lB0[1] + 64 * 32};

    v8f acc[2][4] = {};

    const int KSTEPS = H >> 5;
    cp16(lA0[0], gA0); cp16(lA1[0], gA1);
    cp16(lB0[0], gB0); cp16(lB1[0], gB1);

    for (int k = 0; k < KSTEPS; ++k) {
        const int buf = k & 1;
        cp_wait();
        __syncthreads();
        if (k + 1 < KSTEPS) {
            const int nb = buf ^ 1;
            gA0 += 32; gA1 += 32; gB0 += 32; gB1 += 32;
            cp16(lA0[nb], gA0); cp16(lA1[nb], gA1);
            cp16(lB0[nb], gB0); cp16(lB1[nb], gB1);
        }
        v16bf a0 = load_fragA(As[buf], wm * 32 + (lane & 15), 32, lane);
        v16bf a1 = load_fragA(As[buf], wm * 32 + 16 + (lane & 15), 32, lane);
#pragma unroll
        for (int j = 0; j < 4; ++j) {
            v16bf bb = load_fragB(Bs[buf], wn * 64 + j * 16 + (lane & 15), 32, lane);
            acc[0][j] = WMMA_BF16(a0, bb, acc[0][j]);
            acc[1][j] = WMMA_BF16(a1, bb, acc[1][j]);
        }
    }

    const long long len = srclen[b];
    const int mrow = (lane >> 4) << 3;
    const int ncol = lane & 15;
#pragma unroll
    for (int h = 0; h < 2; ++h) {
#pragma unroll
        for (int j = 0; j < 4; ++j) {
            const int s = s0 + wn * 64 + j * 16 + ncol;
            const bool masked = ((long long)s >= len);
#pragma unroll
            for (int i = 0; i < 8; ++i) {
                const int t = t0 + wm * 32 + h * 16 + mrow + i;
                attn[((size_t)b * T + t) * S + s] = masked ? NEG_INF_F : acc[h][j][i];
            }
        }
    }
}

// ---------------------------------------------------------------------------
// Pass 2: row softmax over S (in place in attn region of d_out)
// ---------------------------------------------------------------------------
__global__ __launch_bounds__(256)
void softmax_kernel(float* __restrict__ attn, int S) {
    const int t = blockIdx.x;
    const int b = blockIdx.y;
    const int T = gridDim.x;
    float* row = attn + ((size_t)b * T + t) * S;
    const int tid = threadIdx.x;
    const int nit = S >> 8;

    __shared__ float red[256];
    float vals[8];

    float m = -3.4e38f;
    for (int i = 0; i < nit && i < 8; ++i) {
        vals[i] = row[tid + (i << 8)];
        m = fmaxf(m, vals[i]);
    }
    red[tid] = m;
    __syncthreads();
    for (int s = 128; s > 0; s >>= 1) {
        if (tid < s) red[tid] = fmaxf(red[tid], red[tid + s]);
        __syncthreads();
    }
    m = red[0];
    __syncthreads();

    float sum = 0.f;
    for (int i = 0; i < nit && i < 8; ++i) {
        vals[i] = __expf(vals[i] - m);
        sum += vals[i];
    }
    red[tid] = sum;
    __syncthreads();
    for (int s = 128; s > 0; s >>= 1) {
        if (tid < s) red[tid] += red[tid + s];
        __syncthreads();
    }
    const float inv = 1.0f / red[0];
    for (int i = 0; i < nit && i < 8; ++i) row[tid + (i << 8)] = vals[i] * inv;
}

// ---------------------------------------------------------------------------
// Pass 3: ctx[t,b,h] = sum_s attn[b,t,s]*enc[s,b,h]; ctx stored bf16 in ws.
// A tile: fp32 attn -> bf16 staging; B tile: encbf transposed [h][s] (pad 40).
// ---------------------------------------------------------------------------
__global__ __launch_bounds__(256)
void ctxgemm_kernel(const float* __restrict__ attn,
                    const unsigned short* __restrict__ encbf,
                    unsigned short* __restrict__ ctxws, int S, int T, int B, int H) {
    const int h0 = blockIdx.x * 128;
    const int t0 = blockIdx.y * 128;
    const int b  = blockIdx.z;
    const int tid  = threadIdx.x;
    const int lane = tid & 31;
    const int wave = tid >> 5;
    const int wm = wave & 3;
    const int wn = wave >> 2;

    __shared__ __align__(16) unsigned short As[2][128 * 32];
    __shared__ __align__(16) unsigned short BsT[2][128 * 40];

    const size_t BH = (size_t)B * H;

    // A staging: 4 chunks of float4 per thread
    const int arow = tid >> 3;            // 0..31, +32,+64,+96
    const int aoff = (tid & 7) << 2;      // 0..28 step 4
    const float* gA = attn + ((size_t)b * T + t0 + arow) * S + aoff;

    // B staging (transpose): 2 chunks of 8 bf16 along h per thread
    const int bs   = tid >> 4;            // s: 0..15 (and +16)
    const int bh   = (tid & 15) << 3;     // h offset: 0..120 step 8
    const unsigned short* gB0 = encbf + (size_t)bs * BH + (size_t)b * H + h0 + bh;
    const unsigned short* gB1 = gB0 + 16 * BH;

    v8f acc[2][4] = {};
    const int KSTEPS = S >> 5;

    for (int k = 0; k < KSTEPS; ++k) {
        const int buf = k & 1;
        __syncthreads();
#pragma unroll
        for (int c = 0; c < 4; ++c) {
            float4 x = *(const float4*)(gA + (size_t)c * 32 * S);
            unsigned short o[4] = {f2bf(x.x), f2bf(x.y), f2bf(x.z), f2bf(x.w)};
            *(uint2*)&As[buf][(arow + c * 32) * 32 + aoff] = *(const uint2*)o;
        }
        {
            uint4 u0 = *(const uint4*)gB0;
            uint4 u1 = *(const uint4*)gB1;
            const unsigned short* p0 = (const unsigned short*)&u0;
            const unsigned short* p1 = (const unsigned short*)&u1;
#pragma unroll
            for (int ii = 0; ii < 8; ++ii) {
                BsT[buf][(bh + ii) * 40 + bs]      = p0[ii];
                BsT[buf][(bh + ii) * 40 + bs + 16] = p1[ii];
            }
        }
        gA += 32;
        gB0 += 32 * BH;
        gB1 += 32 * BH;
        __syncthreads();

        v16bf a0 = load_fragA(As[buf], wm * 32 + (lane & 15), 32, lane);
        v16bf a1 = load_fragA(As[buf], wm * 32 + 16 + (lane & 15), 32, lane);
#pragma unroll
        for (int j = 0; j < 4; ++j) {
            v16bf bb = load_fragB(BsT[buf], wn * 64 + j * 16 + (lane & 15), 40, lane);
            acc[0][j] = WMMA_BF16(a0, bb, acc[0][j]);
            acc[1][j] = WMMA_BF16(a1, bb, acc[1][j]);
        }
    }

    const int mrow = (lane >> 4) << 3;
    const int ncol = lane & 15;
#pragma unroll
    for (int h = 0; h < 2; ++h) {
#pragma unroll
        for (int j = 0; j < 4; ++j) {
            const int hc = h0 + wn * 64 + j * 16 + ncol;
#pragma unroll
            for (int i = 0; i < 8; ++i) {
                const int t = t0 + wm * 32 + h * 16 + mrow + i;
                ctxws[((size_t)t * B + b) * H + hc] = f2bf(acc[h][j][i]);
            }
        }
    }
}

// ---------------------------------------------------------------------------
// Pass 4: states[t,b,h] = tanh([ctx,dec][r,:]. W[h,:] + bias[h]), r = t*B+b
// K = 2H: first H from bf16 ctx ws, second H from bf16 dec. All-async staging.
// ---------------------------------------------------------------------------
__global__ __launch_bounds__(256)
void linear_kernel(const unsigned short* __restrict__ ctxws,
                   const unsigned short* __restrict__ decbf,
                   const unsigned short* __restrict__ Wbf,
                   const float* __restrict__ bias, float* __restrict__ states,
                   int T, int B, int H) {
    const int h0 = blockIdx.x * 128;
    const int r0 = blockIdx.y * 128;
    const int tid  = threadIdx.x;
    const int lane = tid & 31;
    const int wave = tid >> 5;
    const int wm = wave & 3;
    const int wn = wave >> 2;
    const int K2 = 2 * H;

    __shared__ __align__(16) unsigned short As[2][128 * 32];
    __shared__ __align__(16) unsigned short Bs[2][128 * 32];

    const int srow = tid >> 2;
    const int soff = (tid & 3) << 3;

    const unsigned short* aCtx0 = ctxws + (size_t)(r0 + srow) * H + soff;
    const unsigned short* aCtx1 = aCtx0 + (size_t)64 * H;
    const unsigned short* aDec0 = decbf + (size_t)(r0 + srow) * H + soff;
    const unsigned short* aDec1 = aDec0 + (size_t)64 * H;
    const unsigned short* gW0 = Wbf + (size_t)(h0 + srow) * K2 + soff;
    const unsigned short* gW1 = gW0 + (size_t)64 * K2;

    unsigned short* const lA0[2] = {&As[0][srow * 32 + soff], &As[1][srow * 32 + soff]};
    unsigned short* const lA1[2] = {lA0[0] + 64 * 32, lA0[1] + 64 * 32};
    unsigned short* const lB0[2] = {&Bs[0][srow * 32 + soff], &Bs[1][srow * 32 + soff]};
    unsigned short* const lB1[2] = {lB0[0] + 64 * 32, lB0[1] + 64 * 32};

    v8f acc[2][4] = {};
    const int KSTEPS = K2 >> 5;
    const int HSTEPS = H >> 5;

    cp16(lA0[0], aCtx0); cp16(lA1[0], aCtx1);
    cp16(lB0[0], gW0);   cp16(lB1[0], gW1);

    for (int k = 0; k < KSTEPS; ++k) {
        const int buf = k & 1;
        cp_wait();
        __syncthreads();
        if (k + 1 < KSTEPS) {
            const int nb = buf ^ 1;
            const int kn = k + 1;
            const int kof = (kn < HSTEPS) ? (kn << 5) : ((kn - HSTEPS) << 5);
            const unsigned short* a0 = (kn < HSTEPS) ? aCtx0 : aDec0;
            const unsigned short* a1 = (kn < HSTEPS) ? aCtx1 : aDec1;
            cp16(lA0[nb], a0 + kof); cp16(lA1[nb], a1 + kof);
            gW0 += 32; gW1 += 32;
            cp16(lB0[nb], gW0); cp16(lB1[nb], gW1);
        }
        v16bf a0 = load_fragA(As[buf], wm * 32 + (lane & 15), 32, lane);
        v16bf a1 = load_fragA(As[buf], wm * 32 + 16 + (lane & 15), 32, lane);
#pragma unroll
        for (int j = 0; j < 4; ++j) {
            v16bf bb = load_fragB(Bs[buf], wn * 64 + j * 16 + (lane & 15), 32, lane);
            acc[0][j] = WMMA_BF16(a0, bb, acc[0][j]);
            acc[1][j] = WMMA_BF16(a1, bb, acc[1][j]);
        }
    }

    const int mrow = (lane >> 4) << 3;
    const int ncol = lane & 15;
#pragma unroll
    for (int h = 0; h < 2; ++h) {
#pragma unroll
        for (int j = 0; j < 4; ++j) {
            const int hc = h0 + wn * 64 + j * 16 + ncol;
            const float bh = bias[hc];
#pragma unroll
            for (int i = 0; i < 8; ++i) {
                const int r = r0 + wm * 32 + h * 16 + mrow + i;
                states[(size_t)r * H + hc] = tanhf(acc[h][j][i] + bh);
            }
        }
    }
}

extern "C" void kernel_launch(void* const* d_in, const int* in_sizes, int n_in,
                              void* d_out, int out_size, void* d_ws, size_t ws_size,
                              hipStream_t stream) {
    const float*     enc    = (const float*)d_in[0];      // context [S,B,H]
    const long long* srclen = (const long long*)d_in[1];  // [B] int64
    const float*     dec    = (const float*)d_in[2];      // [T,B,H]
    const float*     W      = (const float*)d_in[3];      // [H,2H]
    const float*     bias   = (const float*)d_in[4];      // [H]

    const int B = in_sizes[1];
    const int H = in_sizes[4];
    const int S = (int)((long long)in_sizes[0] / ((long long)B * H));
    const int T = (int)((long long)in_sizes[2] / ((long long)B * H));

    float* states = (float*)d_out;               // [T,B,H]
    float* attn   = states + (size_t)T * B * H;  // [B,T,S]

    const size_t nEnc = (size_t)S * B * H;
    const size_t nDec = (size_t)T * B * H;
    const size_t nW   = (size_t)H * 2 * H;

    unsigned short* encbf = (unsigned short*)d_ws;
    unsigned short* decbf = encbf + nEnc;
    unsigned short* Wbf   = decbf + nDec;
    unsigned short* ctxws = Wbf + nW;

    cvt_bf16_kernel<<<8192, 256, 0, stream>>>(enc, encbf, nEnc);
    cvt_bf16_kernel<<<4096, 256, 0, stream>>>(dec, decbf, nDec);
    cvt_bf16_kernel<<<1024, 256, 0, stream>>>(W, Wbf, nW);

    dim3 g1(S / 128, T / 128, B);
    scores_kernel<<<g1, 256, 0, stream>>>(encbf, decbf, srclen, attn, S, T, B, H);

    dim3 g2(T, B);
    softmax_kernel<<<g2, 256, 0, stream>>>(attn, S);

    dim3 g3(H / 128, T / 128, B);
    ctxgemm_kernel<<<g3, 256, 0, stream>>>(attn, encbf, ctxws, S, T, B, H);

    dim3 g4(H / 128, (T * B) / 128);
    linear_kernel<<<g4, 256, 0, stream>>>(ctxws, decbf, Wbf, bias, states, T, B, H);
}
